// LSTMQNetwork_73907797230071
// MI455X (gfx1250) — compile-verified, matrix operand order checked by reading, not run on hardware
//
#include <hip/hip_runtime.h>
#include <hip/hip_bf16.h>

// ---------------------------------------------------------------------------
// CDNA5 (gfx1250, wave32) WMMA implementation of the tree-LSTM actor/critic.
// All GEMMs run through V_WMMA_F32_16X16X32_F16 (f16 operands, f32 accum).
// Nonlinearities use native gfx1250 transcendental VALU ops (v_exp_f32,
// v_tanh_f32, v_rcp_f32) where available.
// ---------------------------------------------------------------------------

typedef __attribute__((ext_vector_type(16))) _Float16 v16h;
typedef __attribute__((ext_vector_type(8)))  _Float16 v8h;
typedef __attribute__((ext_vector_type(8)))  float    v8f;

#define DEVINL __device__ __forceinline__

#define LOG2E 1.4426950408889634f

DEVINL float fast_exp(float x){
#if __has_builtin(__builtin_amdgcn_exp2f)
  return __builtin_amdgcn_exp2f(x * LOG2E);     // v_exp_f32 (base-2)
#else
  return __expf(x);
#endif
}
DEVINL float fast_rcp(float x){
#if __has_builtin(__builtin_amdgcn_rcpf)
  return __builtin_amdgcn_rcpf(x);              // v_rcp_f32
#else
  return 1.0f / x;
#endif
}
DEVINL float sigmf(float x){ return fast_rcp(1.0f + fast_exp(-x)); }
DEVINL float tanh_fast(float x){
#if __has_builtin(__builtin_amdgcn_tanhf)
  return __builtin_amdgcn_tanhf(x);             // v_tanh_f32
#elif __has_builtin(__builtin_amdgcn_tanh_f32)
  return __builtin_amdgcn_tanh_f32(x);
#else
  // tanh(x) = 1 - 2/(exp(2x)+1), saturates safely via native exp2/rcp
  return 1.0f - 2.0f*fast_rcp(fast_exp(2.0f*x) + 1.0f);
#endif
}
DEVINL float geluf(float x){ return 0.5f*x*(1.0f + erff(x*0.7071067811865475f)); }

// --- WMMA fragment loaders --------------------------------------------------
// wave32: lane = tid&31, half = lane>>4, idx = lane&15.
// A (16x32, MxK): lane idx holds row M=idx; element t in [0,8): K=half*8+t,
//                 t in [8,16): K=16+half*8+(t-8).
// B as row-major W (N,K): lane idx holds W-row (output col) N=idx, same K map.
DEVINL v16h frag_h_masked(const _Float16* rp, int half, int kvalid){
  v16h r;
#pragma unroll
  for(int t=0;t<8;++t){
    int k0 = half*8+t, k1 = 16+half*8+t;
    r[t]   = (k0<kvalid)? rp[k0] : (_Float16)0.0f;
    r[t+8] = (k1<kvalid)? rp[k1] : (_Float16)0.0f;
  }
  return r;
}
DEVINL v16h frag_f32_masked(const float* rp, int half, int kvalid){
  v16h r;
#pragma unroll
  for(int t=0;t<8;++t){
    int k0 = half*8+t, k1 = 16+half*8+t;
    r[t]   = (k0<kvalid)? (_Float16)rp[k0] : (_Float16)0.0f;
    r[t+8] = (k1<kvalid)? (_Float16)rp[k1] : (_Float16)0.0f;
  }
  return r;
}
DEVINL v16h frag_h_full(const _Float16* rp, int half){
  v8h lo = *(const v8h*)(rp + half*8);
  v8h hi = *(const v8h*)(rp + 16 + half*8);
  v16h r;
#pragma unroll
  for(int t=0;t<8;++t){ r[t]=lo[t]; r[t+8]=hi[t]; }
  return r;
}
DEVINL v8f wmma_f16(v16h a, v16h b, v8f c){
  return __builtin_amdgcn_wmma_f32_16x16x32_f16(false, a, false, b,
                                                (short)0, c, false, false);
}

// --- weight conversion ------------------------------------------------------
__global__ void k_f32_to_f16(const float* __restrict__ s,
                             _Float16* __restrict__ d, int n){
  int i = blockIdx.x*blockDim.x + threadIdx.x;
  if(i<n) d[i] = (_Float16)s[i];
}

// --- tree-LSTM: per-level iou/c/h kernel ------------------------------------
// One wave = one 16x16 tile of (rows at this level) x (128 tree-emb cols).
// Keeps 4 f32 accumulators (i, o, u, c-part) sharing A-fragments.
template<int NPT, int PSTART, bool HASCHILD>
__global__ void k_tree_level(const float* __restrict__ forest,
                             const _Float16* __restrict__ w_iou_h,
                             const float* __restrict__ b_iou,
                             const _Float16* __restrict__ u_iou_h,
                             const _Float16* __restrict__ w_c_h,
                             const float* __restrict__ b_c,
                             const _Float16* __restrict__ FC,
                             _Float16* __restrict__ H,
                             float* __restrict__ C,
                             int nTiles)
{
  int wid = (int)((blockIdx.x*blockDim.x + threadIdx.x) >> 5);
  if (wid >= nTiles) return;
  const int lane = threadIdx.x & 31;
  const int half = lane >> 4, idx = lane & 15;
  const int rt = wid >> 3;            // 8 col tiles (128/16)
  const int ct = wid & 7;
  const int colg = ct*16 + idx;       // output column / W-row
  const int row  = rt*16 + idx;       // level-local node row (A operand)
  const int tree = row / NPT;
  const int p    = PSTART + (row - tree*NPT);
  const size_t g    = (size_t)tree*40 + p;
  const size_t gch0 = (size_t)tree*40 + 3*p + 1;

  v8f ai, ao, au, ac;
  {
    float bi = b_iou[colg], bo = b_iou[128+colg], bu = b_iou[256+colg];
    float bc = HASCHILD ? b_c[colg] : 0.0f;
#pragma unroll
    for(int e=0;e<8;++e){ ai[e]=bi; ao[e]=bo; au[e]=bu; ac[e]=bc; }
  }

  // forest @ w_iou.T   (K = 12, zero-padded to 32)
  {
    v16h a  = frag_f32_masked(forest + g*12, half, 12);
    v16h b0 = frag_h_masked(w_iou_h + (size_t)colg*12,       half, 12);
    v16h b1 = frag_h_masked(w_iou_h + (size_t)(128+colg)*12, half, 12);
    v16h b2 = frag_h_masked(w_iou_h + (size_t)(256+colg)*12, half, 12);
    ai = wmma_f16(a,b0,ai); ao = wmma_f16(a,b1,ao); au = wmma_f16(a,b2,au);
  }
  if (HASCHILD){
    const _Float16* hrow  = H  + gch0*128;        // 3 children h, 384 contiguous
    const _Float16* fcrow = FC + (size_t)row*384; // f*c, 384 contiguous
    __builtin_prefetch(hrow, 0, 1);
#pragma unroll
    for(int k0=0;k0<384;k0+=32){
      v16h a  = frag_h_full(hrow + k0, half);
      v16h b0 = frag_h_full(u_iou_h + (size_t)colg*384       + k0, half);
      v16h b1 = frag_h_full(u_iou_h + (size_t)(128+colg)*384 + k0, half);
      v16h b2 = frag_h_full(u_iou_h + (size_t)(256+colg)*384 + k0, half);
      ai = wmma_f16(a,b0,ai); ao = wmma_f16(a,b1,ao); au = wmma_f16(a,b2,au);
      v16h af = frag_h_full(fcrow + k0, half);
      v16h bc = frag_h_full(w_c_h + (size_t)colg*384 + k0, half);
      ac = wmma_f16(af,bc,ac);
    }
  }
  // LSTM nonlinearity epilogue; D tile: lane col = idx, vgpr e row = e+8*half
#pragma unroll
  for(int e=0;e<8;++e){
    int row_e = rt*16 + e + 8*half;
    int tr = row_e / NPT;
    int pe = PSTART + (row_e - tr*NPT);
    size_t ge = (size_t)tr*40 + pe;
    float cv = sigmf(ai[e])*tanh_fast(au[e]) + ac[e];
    float hv = sigmf(ao[e])*tanh_fast(cv);
    C[ge*128 + colg] = cv;
    H[ge*128 + colg] = (_Float16)hv;
  }
}

// --- tree-LSTM: forget-gate * child_c kernel --------------------------------
// Rows = 3 * (#nodes at level): r -> (node m = r/3, child k = r%3).
// f = sigmoid(forest[parent]@w_f.T + b_f + h[child]@u_f.T); FC = f * c[child].
template<int NPT, int PSTART>
__global__ void k_tree_f(const float* __restrict__ forest,
                         const _Float16* __restrict__ w_f_h,
                         const float* __restrict__ b_f,
                         const _Float16* __restrict__ u_f_h,
                         const _Float16* __restrict__ H,
                         const float* __restrict__ C,
                         _Float16* __restrict__ FC,
                         int nTiles)
{
  int wid = (int)((blockIdx.x*blockDim.x + threadIdx.x) >> 5);
  if (wid >= nTiles) return;
  const int lane = threadIdx.x & 31;
  const int half = lane >> 4, idx = lane & 15;
  const int rt = wid >> 3;
  const int ct = wid & 7;
  const int colg = ct*16 + idx;
  const int r = rt*16 + idx;
  const int m = r/3, k = r - 3*m;
  const int tree = m / NPT;
  const int p = PSTART + (m - tree*NPT);
  const size_t gpar = (size_t)tree*40 + p;
  const size_t gch  = (size_t)tree*40 + 3*p + 1 + k;

  v8f acc;
  { float bf = b_f[colg];
#pragma unroll
    for(int e=0;e<8;++e) acc[e]=bf; }

  {
    v16h a = frag_f32_masked(forest + gpar*12, half, 12);
    v16h b = frag_h_masked(w_f_h + (size_t)colg*12, half, 12);
    acc = wmma_f16(a,b,acc);
  }
  const _Float16* hrow = H + gch*128;
#pragma unroll
  for(int k0=0;k0<128;k0+=32){
    v16h a = frag_h_full(hrow + k0, half);
    v16h b = frag_h_full(u_f_h + (size_t)colg*128 + k0, half);
    acc = wmma_f16(a,b,acc);
  }
#pragma unroll
  for(int e=0;e<8;++e){
    int re = rt*16 + e + 8*half;
    int me = re/3, ke = re - 3*me;
    int tr = me / NPT;
    int pe = PSTART + (me - tr*NPT);
    size_t gce = (size_t)tr*40 + 3*pe + 1 + ke;
    float f  = sigmf(acc[e]);
    float cc = C[gce*128 + colg];
    FC[(size_t)me*384 + ke*128 + colg] = (_Float16)(f*cc);
  }
}

// --- generic dense GEMM + bias (+GELU) for the MLP stacks -------------------
template<bool A_F32, bool GELU_F16>
__global__ void k_gemm(const void* __restrict__ Ap,
                       const _Float16* __restrict__ W,
                       const float* __restrict__ bias,
                       void* __restrict__ Outp,
                       int nColTiles, int Nvalid, int K,
                       int ldA, int ldOut, int outOff, int nTiles)
{
  int wid = (int)((blockIdx.x*blockDim.x + threadIdx.x) >> 5);
  if (wid >= nTiles) return;
  const int lane = threadIdx.x & 31;
  const int half = lane >> 4, idx = lane & 15;
  const int rt = wid / nColTiles;
  const int ct = wid - rt*nColTiles;
  const int colg = ct*16 + idx;
  const int wrow = (colg < Nvalid) ? colg : 0;
  const int row  = rt*16 + idx;

  v8f acc;
  { float bv = (colg < Nvalid) ? bias[colg] : 0.0f;
#pragma unroll
    for(int e=0;e<8;++e) acc[e]=bv; }

  const float*    Af = (const float*)Ap;
  const _Float16* Ah = (const _Float16*)Ap;
  for(int k0=0;k0<K;k0+=32){
    int kv = K - k0; if (kv > 32) kv = 32;
    v16h a, b;
    if (!A_F32 && kv==32 && (ldA % 8)==0)
      a = frag_h_full(Ah + (size_t)row*ldA + k0, half);
    else
      a = A_F32 ? frag_f32_masked(Af + (size_t)row*ldA + k0, half, kv)
                : frag_h_masked (Ah + (size_t)row*ldA + k0, half, kv);
    if (kv==32 && (K % 8)==0)
      b = frag_h_full(W + (size_t)wrow*K + k0, half);
    else
      b = frag_h_masked(W + (size_t)wrow*K + k0, half, kv);
    acc = wmma_f16(a, b, acc);
  }
#pragma unroll
  for(int e=0;e<8;++e){
    if (colg < Nvalid){
      int re = rt*16 + e + 8*half;
      float v = acc[e];
      if (GELU_F16)
        ((_Float16*)Outp)[(size_t)re*ldOut + outOff + colg] = (_Float16)geluf(v);
      else
        ((float*)Outp)[(size_t)re*ldOut + outOff + colg] = v;
    }
  }
}

// --- misc small kernels -----------------------------------------------------
__global__ void k_copy_treeemb(const _Float16* __restrict__ H,
                               _Float16* __restrict__ emb){
  int i = blockIdx.x*blockDim.x + threadIdx.x;
  if (i < 2048*128){
    int t = i >> 7, j = i & 127;
    emb[(size_t)t*256 + 128 + j] = H[(size_t)t*40*128 + j]; // root node p=0
  }
}

__global__ void k_softmax5(const float* __restrict__ logits,
                           float* __restrict__ out){
  int t = blockIdx.x*blockDim.x + threadIdx.x;
  if (t < 2048){
    float v[5]; float mx = -3.0e38f;
#pragma unroll
    for(int i=0;i<5;++i){ v[i] = logits[(size_t)t*16 + i]; mx = fmaxf(mx, v[i]); }
    float s = 0.0f;
#pragma unroll
    for(int i=0;i<5;++i){ v[i] = fast_exp(v[i]-mx); s += v[i]; }
    float inv = fast_rcp(s);
#pragma unroll
    for(int i=0;i<5;++i) out[(size_t)t*5 + i] = v[i]*inv;
  }
}

__global__ void k_critic_dot(const _Float16* __restrict__ v2,
                             const float* __restrict__ w2,
                             const float* __restrict__ b2,
                             float* __restrict__ cv){
  int wid = (int)((blockIdx.x*blockDim.x + threadIdx.x) >> 5);
  int lane = threadIdx.x & 31;
  if (wid >= 2048) return;
  float s = 0.0f;
#pragma unroll
  for(int c0=0;c0<128;c0+=32) s += (float)v2[(size_t)wid*128 + c0 + lane] * w2[c0 + lane];
  for(int o=16;o>0;o>>=1) s += __shfl_xor(s, o, 32);
  if (lane==0) cv[wid] = s + b2[0];
}

__global__ void k_critic_mean(const float* __restrict__ cv,
                              float* __restrict__ out){
  int b = blockIdx.x*blockDim.x + threadIdx.x;
  if (b < 32){
    float s = 0.0f;
    for(int a=0;a<64;++a) s += cv[b*64 + a];
    out[10240 + b] = s * (1.0f/64.0f);
  }
}

// ---------------------------------------------------------------------------
extern "C" void kernel_launch(void* const* d_in, const int* in_sizes, int n_in,
                              void* d_out, int out_size, void* d_ws, size_t ws_size,
                              hipStream_t stream)
{
  (void)in_sizes; (void)n_in; (void)out_size; (void)ws_size;
  const float* agents = (const float*)d_in[0];   // (2048, 83)
  const float* forest = (const float*)d_in[1];   // (81920, 12)
  // d_in[2..4]: adjacency/node_order/edge_order -- fixed topology, unused.
  const float* w_iou = (const float*)d_in[5];
  const float* b_iou = (const float*)d_in[6];
  const float* u_iou = (const float*)d_in[7];
  const float* w_c   = (const float*)d_in[8];
  const float* b_c   = (const float*)d_in[9];
  const float* w_f   = (const float*)d_in[10];
  const float* b_f   = (const float*)d_in[11];
  const float* u_f   = (const float*)d_in[12];
  const float* aw0 = (const float*)d_in[13]; const float* ab0 = (const float*)d_in[14];
  const float* aw1 = (const float*)d_in[15]; const float* ab1 = (const float*)d_in[16];
  const float* aw2 = (const float*)d_in[17]; const float* ab2 = (const float*)d_in[18];
  const float* aw3 = (const float*)d_in[19]; const float* ab3 = (const float*)d_in[20];
  const float* pw0 = (const float*)d_in[21]; const float* pb0 = (const float*)d_in[22];
  const float* pw1 = (const float*)d_in[23]; const float* pb1 = (const float*)d_in[24];
  const float* pw2 = (const float*)d_in[25]; const float* pb2 = (const float*)d_in[26];
  const float* vw0 = (const float*)d_in[27]; const float* vb0 = (const float*)d_in[28];
  const float* vw1 = (const float*)d_in[29]; const float* vb1 = (const float*)d_in[30];
  const float* vw2 = (const float*)d_in[31]; const float* vb2 = (const float*)d_in[32];

  char* ws = (char*)d_ws; size_t off = 0;
  auto alloc = [&](size_t bytes)->void*{
    void* p = ws + off; off = (off + bytes + 255) & ~(size_t)255; return p;
  };
  _Float16* H  = (_Float16*)alloc((size_t)81920*128*2);
  float*    C  = (float*)   alloc((size_t)81920*128*4);
  _Float16* FC = (_Float16*)alloc((size_t)18432*384*2);
  _Float16* w_iou_h = (_Float16*)alloc((size_t)4608*2);
  _Float16* w_f_h   = (_Float16*)alloc((size_t)1536*2);
  _Float16* u_iou_h = (_Float16*)alloc((size_t)147456*2);
  _Float16* u_f_h   = (_Float16*)alloc((size_t)16384*2);
  _Float16* w_c_h   = (_Float16*)alloc((size_t)49152*2);
  _Float16* aw0_h = (_Float16*)alloc((size_t)21248*2);
  _Float16* aw1_h = (_Float16*)alloc((size_t)65536*2);
  _Float16* aw2_h = (_Float16*)alloc((size_t)65536*2);
  _Float16* aw3_h = (_Float16*)alloc((size_t)32768*2);
  _Float16* pw0_h = (_Float16*)alloc((size_t)65536*2);
  _Float16* pw1_h = (_Float16*)alloc((size_t)32768*2);
  _Float16* pw2_h = (_Float16*)alloc((size_t)640*2);
  _Float16* vw0_h = (_Float16*)alloc((size_t)65536*2);
  _Float16* vw1_h = (_Float16*)alloc((size_t)32768*2);
  _Float16* x1  = (_Float16*)alloc((size_t)2048*256*2);
  _Float16* x2  = (_Float16*)alloc((size_t)2048*256*2);
  _Float16* emb = (_Float16*)alloc((size_t)2048*256*2);
  _Float16* a1  = (_Float16*)alloc((size_t)2048*256*2);
  _Float16* a2  = (_Float16*)alloc((size_t)2048*128*2);
  _Float16* v1  = (_Float16*)alloc((size_t)2048*256*2);
  _Float16* v2  = (_Float16*)alloc((size_t)2048*128*2);
  float* logits = (float*)alloc((size_t)2048*16*4);
  float* cv     = (float*)alloc((size_t)2048*4);

  auto cvt = [&](const float* s, _Float16* d, int n){
    k_f32_to_f16<<<(n+255)/256, 256, 0, stream>>>(s, d, n);
  };
  cvt(w_iou, w_iou_h, 4608);   cvt(w_f, w_f_h, 1536);
  cvt(u_iou, u_iou_h, 147456); cvt(u_f, u_f_h, 16384);
  cvt(w_c, w_c_h, 49152);
  cvt(aw0, aw0_h, 21248); cvt(aw1, aw1_h, 65536);
  cvt(aw2, aw2_h, 65536); cvt(aw3, aw3_h, 32768);
  cvt(pw0, pw0_h, 65536); cvt(pw1, pw1_h, 32768); cvt(pw2, pw2_h, 640);
  cvt(vw0, vw0_h, 65536); cvt(vw1, vw1_h, 32768);

  const int WPB = 4, TPB = WPB*32;
  auto blk = [&](int nt){ return (nt + WPB - 1)/WPB; };

  // ---- tree LSTM levels (leaves first) ----
  { int nt = (2048*27/16)*8;   // level 0: 27 leaves/tree
    k_tree_level<27,13,false><<<blk(nt), TPB, 0, stream>>>(
        forest, w_iou_h, b_iou, u_iou_h, w_c_h, b_c, FC, H, C, nt); }
  { int ntf = (2048*9*3/16)*8;
    k_tree_f<9,4><<<blk(ntf), TPB, 0, stream>>>(forest, w_f_h, b_f, u_f_h, H, C, FC, ntf);
    int nt = (2048*9/16)*8;
    k_tree_level<9,4,true><<<blk(nt), TPB, 0, stream>>>(
        forest, w_iou_h, b_iou, u_iou_h, w_c_h, b_c, FC, H, C, nt); }
  { int ntf = (2048*3*3/16)*8;
    k_tree_f<3,1><<<blk(ntf), TPB, 0, stream>>>(forest, w_f_h, b_f, u_f_h, H, C, FC, ntf);
    int nt = (2048*3/16)*8;
    k_tree_level<3,1,true><<<blk(nt), TPB, 0, stream>>>(
        forest, w_iou_h, b_iou, u_iou_h, w_c_h, b_c, FC, H, C, nt); }
  { int ntf = (2048*3/16)*8;
    k_tree_f<1,0><<<blk(ntf), TPB, 0, stream>>>(forest, w_f_h, b_f, u_f_h, H, C, FC, ntf);
    int nt = (2048/16)*8;
    k_tree_level<1,0,true><<<blk(nt), TPB, 0, stream>>>(
        forest, w_iou_h, b_iou, u_iou_h, w_c_h, b_c, FC, H, C, nt); }

  // ---- attr MLP: 83 -> 256 -> 256 -> 256 -> 128 (GELU) ----
  { int nt = 128*16; k_gemm<true ,true ><<<blk(nt), TPB, 0, stream>>>(
        agents, aw0_h, ab0, x1, 16, 256,  83,  83, 256, 0, nt); }
  { int nt = 128*16; k_gemm<false,true ><<<blk(nt), TPB, 0, stream>>>(
        x1, aw1_h, ab1, x2, 16, 256, 256, 256, 256, 0, nt); }
  { int nt = 128*16; k_gemm<false,true ><<<blk(nt), TPB, 0, stream>>>(
        x2, aw2_h, ab2, x1, 16, 256, 256, 256, 256, 0, nt); }
  { int nt = 128*8;  k_gemm<false,true ><<<blk(nt), TPB, 0, stream>>>(
        x1, aw3_h, ab3, emb, 8, 128, 256, 256, 256, 0, nt); }
  k_copy_treeemb<<<(2048*128+255)/256, 256, 0, stream>>>(H, emb);

  // ---- actor head: 256 -> 256 -> 128 -> 5 (softmax) ----
  { int nt = 128*16; k_gemm<false,true ><<<blk(nt), TPB, 0, stream>>>(
        emb, pw0_h, pb0, a1, 16, 256, 256, 256, 256, 0, nt); }
  { int nt = 128*8;  k_gemm<false,true ><<<blk(nt), TPB, 0, stream>>>(
        a1, pw1_h, pb1, a2, 8, 128, 256, 256, 128, 0, nt); }
  { int nt = 128*1;  k_gemm<false,false><<<blk(nt), TPB, 0, stream>>>(
        a2, pw2_h, pb2, logits, 1, 5, 128, 128, 16, 0, nt); }
  k_softmax5<<<(2048+255)/256, 256, 0, stream>>>(logits, (float*)d_out);

  // ---- critic head: 256 -> 256 -> 128 -> 1, mean over agents ----
  { int nt = 128*16; k_gemm<false,true ><<<blk(nt), TPB, 0, stream>>>(
        emb, vw0_h, vb0, v1, 16, 256, 256, 256, 256, 0, nt); }
  { int nt = 128*8;  k_gemm<false,true ><<<blk(nt), TPB, 0, stream>>>(
        v1, vw1_h, vb1, v2, 8, 128, 256, 256, 128, 0, nt); }
  k_critic_dot<<<(2048*32 + 255)/256, 256, 0, stream>>>(v2, vw2, vb2, cv);
  k_critic_mean<<<1, 32, 0, stream>>>(cv, (float*)d_out);
}
